// AttentionalLSTM_53042846105688
// MI455X (gfx1250) — matrix operand order repaired and checked by reading, not compiled, not run on hardware
//
#include <hip/hip_runtime.h>
#include <hip/hip_bf16.h>
#include <math.h>

// ---------------------------------------------------------------------------
// Attentional LSTM for MI455X (gfx1250, wave32, WMMA).
// B=64, T=1024, D=512, H=512 hard-coded from the reference setup.
// ---------------------------------------------------------------------------

#define BB 64
#define TT 1024
#define DD 512
#define HH 512
#define G4 (4 * HH)   // 2048

typedef __attribute__((ext_vector_type(16))) __bf16 v16bf;
typedef __attribute__((ext_vector_type(8)))  float  v8f;

union FragBF {
    v16bf v;
    uint4 q[2];
    unsigned short s[16];
};
union FragF {
    v8f   v;
    float f[8];
};

__device__ __forceinline__ unsigned short f2bf(float f) {
    union { float f; unsigned u; } x;
    x.f = f;
    unsigned r = x.u + 0x7FFFu + ((x.u >> 16) & 1u);   // round-to-nearest-even
    return (unsigned short)(r >> 16);
}

__device__ __forceinline__ float sigmoidf(float x) {
    return 1.0f / (1.0f + __expf(-x));
}

// ---------------------------------------------------------------------------
// One-time prep: convert weights to bf16 [N,K] row-major, fuse biases,
// init h (bf16) and c (fp32) state.
// ---------------------------------------------------------------------------
__global__ void prep_kernel(const float* __restrict__ W_ih, const float* __restrict__ W_hh,
                            const float* __restrict__ b_ih, const float* __restrict__ b_hh,
                            const float* __restrict__ W_in, const float* __restrict__ W_out,
                            const float* __restrict__ h0,   const float* __restrict__ c0,
                            unsigned short* __restrict__ Wg,   float* __restrict__ biasg,
                            unsigned short* __restrict__ Winb, unsigned short* __restrict__ Woutb,
                            unsigned short* __restrict__ hbf,  float* __restrict__ cstate)
{
    const int NG = G4 * (DD + HH);   // 2048*1024  gate weights (x|h concat along K)
    const int NB = G4;               // fused bias
    const int NI = HH * HH;          // W_in
    const int NO = HH * 2 * HH;      // W_out
    const int NH = BB * HH;          // h / c state
    const int total = NG + NB + NI + NO + NH + NH;

    for (int idx = blockIdx.x * blockDim.x + threadIdx.x; idx < total;
         idx += gridDim.x * blockDim.x) {
        int i = idx;
        if (i < NG) {
            int n = i >> 10, k = i & 1023;
            float v = (k < DD) ? W_ih[n * DD + k] : W_hh[n * HH + (k - DD)];
            Wg[i] = f2bf(v);
        } else if ((i -= NG) < NB) {
            biasg[i] = b_ih[i] + b_hh[i];
        } else if ((i -= NB) < NI) {
            Winb[i] = f2bf(W_in[i]);
        } else if ((i -= NI) < NO) {
            Woutb[i] = f2bf(W_out[i]);
        } else if ((i -= NO) < NH) {
            hbf[i] = f2bf(h0[i]);
        } else {
            i -= NH;
            cstate[i] = c0[i];
        }
    }
}

// ---------------------------------------------------------------------------
// Per-step: convert x_t = input[:, t, :] to bf16 [B, D] row-major.
// ---------------------------------------------------------------------------
__global__ void convert_x_kernel(const float* __restrict__ input, int t,
                                 unsigned short* __restrict__ xbf)
{
    int idx = blockIdx.x * blockDim.x + threadIdx.x;   // 0 .. B*D-1
    int m = idx >> 9, k = idx & 511;
    xbf[idx] = f2bf(input[((size_t)m * TT + t) * DD + k]);
}

// ---------------------------------------------------------------------------
// Generic bf16 WMMA GEMM:  C[M,N] = concat(A0,A1)[M,K] * W[N,K]^T (+bias)(tanh?)
// A0: [M, K0] bf16 row-major; A1: [M, K-K0] bf16 row-major (may be null).
// W : [N, K]  bf16 row-major (so B fragments are contiguous 16B loads).
// One wave per 16x64 C strip: 4 independent accumulators interleave 4
// WMMAs per k-chunk (hides the 5-slot WMMA->WMMA RAW hazard) and give 4x
// A-fragment reuse.  K is consumed in two straight-line phases (A0, A1) so
// the hot loop has no address selects.
// ---------------------------------------------------------------------------
__global__ void wmma_gemm_kernel(const unsigned short* __restrict__ A0,
                                 const unsigned short* __restrict__ A1, int K0,
                                 const unsigned short* __restrict__ W,
                                 const float* __restrict__ bias,
                                 float* __restrict__ C, int ldc,
                                 unsigned short* __restrict__ Cbf,
                                 int M, int N, int K, int act)
{
    const int wave   = (blockIdx.x * blockDim.x + threadIdx.x) >> 5;
    const int lane   = threadIdx.x & 31;
    const int strips = N >> 6;                 // N-tiles of 64
    const int mTile  = wave / strips;
    const int n4     = wave % strips;
    if (mTile * 16 >= M) return;

    const int row  = mTile * 16 + (lane & 15);
    const int col  = n4 * 64 + (lane & 15);    // column of first N-tile
    const int hiA  = (lane & 16) ? 16 : 0;     // A K-offset for upper half-wave
    const int hiB  = (lane & 16) ? 16 : 0;     // B K-offset for upper half-wave

    // W rows for the 4 N-tiles of this strip
    const unsigned short* wr0 = W + (size_t)col * K;
    const unsigned short* wr1 = wr0 + (size_t)16 * K;
    const unsigned short* wr2 = wr0 + (size_t)32 * K;
    const unsigned short* wr3 = wr0 + (size_t)48 * K;

    FragF acc0, acc1, acc2, acc3;
#pragma unroll
    for (int r = 0; r < 8; ++r) {
        acc0.f[r] = 0.0f; acc1.f[r] = 0.0f; acc2.f[r] = 0.0f; acc3.f[r] = 0.0f;
    }

    // ---- phase 1: A0, K in [0, K0) ----
    {
        const unsigned short* ar = A0 + (size_t)row * K0 + hiA;
        for (int kc = 0; kc < K0; kc += 32) {
            FragBF a, b0, b1, b2, b3;
            a.q[0] = *(const uint4*)(ar + kc);
            a.q[1] = *(const uint4*)(ar + kc + 8);
            const int kW = kc + hiB;
            b0.q[0] = *(const uint4*)(wr0 + kW); b0.q[1] = *(const uint4*)(wr0 + kW + 8);
            b1.q[0] = *(const uint4*)(wr1 + kW); b1.q[1] = *(const uint4*)(wr1 + kW + 8);
            b2.q[0] = *(const uint4*)(wr2 + kW); b2.q[1] = *(const uint4*)(wr2 + kW + 8);
            b3.q[0] = *(const uint4*)(wr3 + kW); b3.q[1] = *(const uint4*)(wr3 + kW + 8);
            acc0.v = __builtin_amdgcn_wmma_f32_16x16x32_bf16(false, a.v, false, b0.v, (short)0, acc0.v, false, false);
            acc1.v = __builtin_amdgcn_wmma_f32_16x16x32_bf16(false, a.v, false, b1.v, (short)0, acc1.v, false, false);
            acc2.v = __builtin_amdgcn_wmma_f32_16x16x32_bf16(false, a.v, false, b2.v, (short)0, acc2.v, false, false);
            acc3.v = __builtin_amdgcn_wmma_f32_16x16x32_bf16(false, a.v, false, b3.v, (short)0, acc3.v, false, false);
        }
    }
    // ---- phase 2: A1, K in [K0, K) ----
    if (A1) {
        const int K1 = K - K0;
        const unsigned short* ar = A1 + (size_t)row * K1 + hiA;
        for (int kc = 0; kc < K1; kc += 32) {
            FragBF a, b0, b1, b2, b3;
            a.q[0] = *(const uint4*)(ar + kc);
            a.q[1] = *(const uint4*)(ar + kc + 8);
            const int kW = K0 + kc + hiB;
            b0.q[0] = *(const uint4*)(wr0 + kW); b0.q[1] = *(const uint4*)(wr0 + kW + 8);
            b1.q[0] = *(const uint4*)(wr1 + kW); b1.q[1] = *(const uint4*)(wr1 + kW + 8);
            b2.q[0] = *(const uint4*)(wr2 + kW); b2.q[1] = *(const uint4*)(wr2 + kW + 8);
            b3.q[0] = *(const uint4*)(wr3 + kW); b3.q[1] = *(const uint4*)(wr3 + kW + 8);
            acc0.v = __builtin_amdgcn_wmma_f32_16x16x32_bf16(false, a.v, false, b0.v, (short)0, acc0.v, false, false);
            acc1.v = __builtin_amdgcn_wmma_f32_16x16x32_bf16(false, a.v, false, b1.v, (short)0, acc1.v, false, false);
            acc2.v = __builtin_amdgcn_wmma_f32_16x16x32_bf16(false, a.v, false, b2.v, (short)0, acc2.v, false, false);
            acc3.v = __builtin_amdgcn_wmma_f32_16x16x32_bf16(false, a.v, false, b3.v, (short)0, acc3.v, false, false);
        }
    }

    // ---- epilogue: C layout lane = col(n), VGPR r = row r + 8*(lane>=16) ----
    const int mbase = mTile * 16 + ((lane & 16) ? 8 : 0);
    const FragF* accs[4] = { &acc0, &acc1, &acc2, &acc3 };
#pragma unroll
    for (int j = 0; j < 4; ++j) {
        const int cj = col + 16 * j;
        const float bv = bias ? bias[cj] : 0.0f;
#pragma unroll
        for (int r = 0; r < 8; ++r) {
            float v = accs[j]->f[r] + bv;
            if (act) v = tanhf(v);
            C[(size_t)(mbase + r) * ldc + cj] = v;
            if (Cbf) Cbf[(mbase + r) * N + cj] = f2bf(v);
        }
    }
}

// ---------------------------------------------------------------------------
// LSTM pointwise cell: gates [B, 4H] (i,f,g,o), c state fp32, hy -> bf16.
// ---------------------------------------------------------------------------
__global__ void lstm_cell_kernel(const float* __restrict__ gates,
                                 float* __restrict__ cstate,
                                 unsigned short* __restrict__ hybf)
{
    int idx = blockIdx.x * blockDim.x + threadIdx.x;   // 0 .. B*H-1
    int m = idx >> 9, u = idx & 511;
    const float* g = gates + (size_t)m * G4;
    float gi = g[u];
    float gf = g[HH + u];
    float gg = g[2 * HH + u];
    float go = g[3 * HH + u];
    float c  = cstate[idx];
    float cy = sigmoidf(gf) * c + sigmoidf(gi) * tanhf(gg);
    float hy = sigmoidf(go) * tanhf(cy);
    cstate[idx] = cy;
    hybf[idx]   = f2bf(hy);
}

// ---------------------------------------------------------------------------
// Attention: scores = ctx[b] . q[b] (masked), softmax over L=1024,
// wc[b] = sum_l probs * ctx[b,l,:].  One block (256 thr) per batch row.
// ctx (134 MB fp32) is L2-resident on MI455X (192 MB L2); prefetch keeps
// the WGP$ fed along the row walk (lowers to global_prefetch_b8).
// ---------------------------------------------------------------------------
__global__ void attention_kernel(const float* __restrict__ ctx,
                                 const unsigned char* __restrict__ mask,
                                 const float* __restrict__ q,
                                 unsigned short* __restrict__ wcbf)
{
    __shared__ float qsh[HH];
    __shared__ float ez[TT];
    __shared__ float red[256];

    const int b   = blockIdx.x;
    const int tid = threadIdx.x;

    qsh[tid]       = q[b * HH + tid];
    qsh[tid + 256] = q[b * HH + 256 + tid];
    __syncthreads();

    // ---- scores: 4 rows of ctx per thread, fp32 dot over H=512
    float sc[4];
#pragma unroll
    for (int j = 0; j < 4; ++j) {
        int l = tid + j * 256;
        const float* cp = ctx + ((size_t)b * TT + l) * HH;
        __builtin_prefetch(cp, 0, 0);
        float acc = 0.0f;
        for (int h = 0; h < HH; h += 4) {
            float4 v = *(const float4*)(cp + h);
            acc += v.x * qsh[h] + v.y * qsh[h + 1] + v.z * qsh[h + 2] + v.w * qsh[h + 3];
        }
        if (mask[b * TT + l]) acc = -__builtin_inff();
        sc[j] = acc;
    }

    // ---- softmax: max-reduce
    float mx = fmaxf(fmaxf(sc[0], sc[1]), fmaxf(sc[2], sc[3]));
    red[tid] = mx;
    __syncthreads();
    for (int off = 128; off > 0; off >>= 1) {
        if (tid < off) red[tid] = fmaxf(red[tid], red[tid + off]);
        __syncthreads();
    }
    mx = red[0];
    __syncthreads();

    // ---- exp + sum-reduce (unnormalized e stored; inverse folded into wc)
    float s = 0.0f;
#pragma unroll
    for (int j = 0; j < 4; ++j) {
        float e = __expf(sc[j] - mx);
        ez[tid + j * 256] = e;
        s += e;
    }
    red[tid] = s;
    __syncthreads();
    for (int off = 128; off > 0; off >>= 1) {
        if (tid < off) red[tid] += red[tid + off];
        __syncthreads();
    }
    const float inv = 1.0f / red[0];
    __syncthreads();

    // ---- weighted context: 2 output dims per thread, coalesced ctx columns
    float w0 = 0.0f, w1 = 0.0f;
    for (int l = 0; l < TT; ++l) {
        float p = ez[l];
        const float* cp = ctx + ((size_t)b * TT + l) * HH;
        __builtin_prefetch(cp + 8 * HH, 0, 0);   // 8 rows ahead
        w0 += p * cp[tid];
        w1 += p * cp[tid + 256];
    }
    wcbf[b * HH + tid]       = f2bf(w0 * inv);
    wcbf[b * HH + 256 + tid] = f2bf(w1 * inv);
}

// ---------------------------------------------------------------------------
// Finalize: h_fin = output[:, T-1, :], c_fin = cstate.
// ---------------------------------------------------------------------------
__global__ void finalize_kernel(float* __restrict__ out, const float* __restrict__ cstate)
{
    int idx = blockIdx.x * blockDim.x + threadIdx.x;   // 0 .. B*H-1
    int m = idx >> 9, u = idx & 511;
    const size_t BTH = (size_t)BB * TT * HH;
    out[BTH + idx]           = out[(size_t)m * TT * HH + (size_t)(TT - 1) * HH + u];
    out[BTH + BB * HH + idx] = cstate[idx];
}

// ---------------------------------------------------------------------------
// Host launch: graph-capturable, deterministic, stream-ordered only.
// ---------------------------------------------------------------------------
extern "C" void kernel_launch(void* const* d_in, const int* in_sizes, int n_in,
                              void* d_out, int out_size, void* d_ws, size_t ws_size,
                              hipStream_t stream)
{
    const float* input = (const float*)d_in[0];   // [B,T,D]
    const float* h0    = (const float*)d_in[1];   // [B,H]
    const float* c0    = (const float*)d_in[2];   // [B,H]
    const float* ctx   = (const float*)d_in[3];   // [B,T,H]
    const unsigned char* mask = (const unsigned char*)d_in[4];  // [B,T] bool
    const float* W_ih  = (const float*)d_in[5];   // [4H,D]
    const float* b_ih  = (const float*)d_in[6];   // [4H]
    const float* W_hh  = (const float*)d_in[7];   // [4H,H]
    const float* b_hh  = (const float*)d_in[8];   // [4H]
    const float* W_in  = (const float*)d_in[9];   // [H,H]
    const float* W_out = (const float*)d_in[10];  // [H,2H]
    float* out = (float*)d_out;                   // [B,T,H] ++ h_fin ++ c_fin

    // ---- workspace carve (all 256B aligned; total ~6.5 MB)
    char* p = (char*)d_ws;
    auto carve = [&](size_t bytes) -> void* {
        void* r = (void*)p;
        p += (bytes + 255) & ~(size_t)255;
        return r;
    };
    unsigned short* Wg    = (unsigned short*)carve((size_t)G4 * (DD + HH) * 2);
    float*          biasg = (float*)         carve((size_t)G4 * 4);
    unsigned short* Winb  = (unsigned short*)carve((size_t)HH * HH * 2);
    unsigned short* Woutb = (unsigned short*)carve((size_t)HH * 2 * HH * 2);
    unsigned short* xbf   = (unsigned short*)carve((size_t)BB * DD * 2);
    unsigned short* hbf   = (unsigned short*)carve((size_t)BB * HH * 2);
    unsigned short* hybf  = (unsigned short*)carve((size_t)BB * HH * 2);
    unsigned short* wcbf  = (unsigned short*)carve((size_t)BB * HH * 2);
    float*          cst   = (float*)         carve((size_t)BB * HH * 4);
    float*          gates = (float*)         carve((size_t)BB * G4 * 4);
    float*          qbuf  = (float*)         carve((size_t)BB * HH * 4);

    prep_kernel<<<2048, 256, 0, stream>>>(W_ih, W_hh, b_ih, b_hh, W_in, W_out, h0, c0,
                                          Wg, biasg, Winb, Woutb, hbf, cst);

    const int ldOut = TT * HH;   // row stride of output [B,T,H] along B

    // wave counts: one wave per 16x64 C strip
    const int gatesWaves = (BB / 16) * (G4 / 64);  // 4*32 = 128
    const int smallWaves = (BB / 16) * (HH / 64);  // 4*8  = 32

    for (int t = 0; t < TT; ++t) {
        convert_x_kernel<<<(BB * DD) / 256, 256, 0, stream>>>(input, t, xbf);

        // gates = [x_t | h] * Wg^T + bias  : M=64, N=2048, K=1024
        wmma_gemm_kernel<<<gatesWaves / 8, 256, 0, stream>>>(
            xbf, hbf, DD, Wg, biasg, gates, G4, nullptr, BB, G4, DD + HH, 0);

        lstm_cell_kernel<<<(BB * HH) / 256, 256, 0, stream>>>(gates, cst, hybf);

        // q = hy * W_in^T : M=64, N=512, K=512
        wmma_gemm_kernel<<<smallWaves / 8, 256, 0, stream>>>(
            hybf, nullptr, HH, Winb, nullptr, qbuf, HH, nullptr, BB, HH, HH, 0);

        attention_kernel<<<BB, 256, 0, stream>>>(ctx, mask, qbuf, wcbf);

        // h_tilde = tanh([wc | hy] * W_out^T) -> output[:,t,:] and next h (bf16)
        wmma_gemm_kernel<<<smallWaves / 8, 256, 0, stream>>>(
            wcbf, hybf, HH, Woutb, nullptr, out + (size_t)t * HH, ldOut, hbf,
            BB, HH, 2 * HH, 1);
    }

    finalize_kernel<<<(BB * HH) / 256, 256, 0, stream>>>(out, cst);
}
